// SoftMoE_28656021799162
// MI455X (gfx1250) — matrix operand order repaired and checked by reading, not compiled, not run on hardware
//
#include <hip/hip_runtime.h>
#include <hip/hip_bf16.h>
#include <math.h>

typedef __attribute__((ext_vector_type(16))) __bf16 v16bf;
typedef __attribute__((ext_vector_type(8)))  float  v8f;

#define Dm   1024
#define Em   8
#define Km   2
#define FFm  4096
#define Nm   4096            // B*T tokens
#define MT   32              // tokens per tile
#define MAX_TILES (Nm/MT)    // 128 tiles per expert worst case

// ---------------------------------------------------------------------------
// 1) Router: one wave32 per token. logits = x . Wr^T, top-2, softmax.
// ---------------------------------------------------------------------------
__global__ __launch_bounds__(256) void moe_router(
    const float* __restrict__ x, const float* __restrict__ Wr,
    int* __restrict__ idx, float* __restrict__ wts) {
  int wid  = threadIdx.x >> 5;
  int lane = threadIdx.x & 31;
  int n = blockIdx.x * 8 + wid;
  if (n >= Nm) return;

  const float* xr = x + (size_t)n * Dm;
  float xv[32];
#pragma unroll
  for (int j = 0; j < 32; ++j) xv[j] = xr[lane + 32 * j];

  float logit[Em];
#pragma unroll
  for (int e = 0; e < Em; ++e) {
    const float* wr = Wr + e * Dm;
    float acc = 0.f;
#pragma unroll
    for (int j = 0; j < 32; ++j) acc += xv[j] * wr[lane + 32 * j];
#pragma unroll
    for (int off = 16; off > 0; off >>= 1) acc += __shfl_xor(acc, off, 32);
    logit[e] = acc;   // full sum now in every lane
  }

  if (lane == 0) {
    int e0 = 0; float s0 = logit[0];
#pragma unroll
    for (int e = 1; e < Em; ++e) if (logit[e] > s0) { s0 = logit[e]; e0 = e; }
    int e1 = -1; float s1 = -3.4e38f;
#pragma unroll
    for (int e = 0; e < Em; ++e)
      if (e != e0 && logit[e] > s1) { s1 = logit[e]; e1 = e; }
    // softmax over {s0, s1}, s0 >= s1
    float t  = expf(s1 - s0);
    float w0 = 1.0f / (1.0f + t);
    idx[n * 2 + 0] = e0; idx[n * 2 + 1] = e1;
    wts[n * 2 + 0] = w0; wts[n * 2 + 1] = t * w0;
  }
}

// ---------------------------------------------------------------------------
// 2) Zero per-expert counters
// ---------------------------------------------------------------------------
__global__ void moe_zero_counts(int* __restrict__ counts) {
  if (threadIdx.x < Em) counts[threadIdx.x] = 0;
}

// ---------------------------------------------------------------------------
// 3) Scatter assignments into per-expert token lists (token | slot<<16)
// ---------------------------------------------------------------------------
__global__ void moe_scatter(const int* __restrict__ idx,
                            int* __restrict__ counts, int* __restrict__ lists) {
  int i = blockIdx.x * blockDim.x + threadIdx.x;
  if (i >= Nm * Km) return;
  int e = idx[i];
  int pos = atomicAdd(&counts[e], 1);
  int n = i >> 1, slot = i & 1;
  lists[e * Nm + pos] = n | (slot << 16);
}

// ---------------------------------------------------------------------------
// 4) Fused expert FFN: per block, 32 tokens of one expert (two 16-row WMMA
//    tiles share every B fragment). bf16 WMMA, fp32 accumulate, exact GELU.
// ---------------------------------------------------------------------------
__global__ __launch_bounds__(256) void moe_expert_ffn(
    const float* __restrict__ x,  const float* __restrict__ W1,
    const float* __restrict__ b1, const float* __restrict__ W2,
    const float* __restrict__ b2, const int* __restrict__ counts,
    const int* __restrict__ lists, float* __restrict__ yws) {
  __shared__ uint4 xs4[MT * Dm * 2 / 16];   // 32 x 1024 bf16 (64 KB)
  __shared__ uint4 hs4[MT * 128 * 2 / 16];  // 32 x 128  bf16 (8 KB)
  __shared__ int s_tok[MT], s_slot[MT], s_valid[MT];
  __bf16* xs = (__bf16*)xs4;
  __bf16* hs = (__bf16*)hs4;

  int e    = blockIdx.x / MAX_TILES;
  int tile = blockIdx.x % MAX_TILES;
  int count = counts[e];
  if (tile * MT >= count) return;   // uniform exit for the whole block

  int tid = threadIdx.x;
  if (tid < MT) {
    int r = tile * MT + tid;
    if (r < count) {
      int v = lists[e * Nm + r];
      s_tok[tid] = v & 0xFFFF; s_slot[tid] = v >> 16; s_valid[tid] = 1;
    } else {
      s_tok[tid] = 0; s_slot[tid] = 0; s_valid[tid] = 0;
    }
  }
  __syncthreads();

  // Gather x rows -> LDS as bf16. thread: row = tid/8, 128 contiguous cols.
  {
    int r  = tid >> 3;
    int c0 = (tid & 7) * 128;
    const float* xr = x + (size_t)s_tok[r] * Dm + c0;
#pragma unroll
    for (int c = 0; c < 128; c += 4) {
      float4 f = *(const float4*)(xr + c);
      xs[r * Dm + c0 + c + 0] = (__bf16)f.x;
      xs[r * Dm + c0 + c + 1] = (__bf16)f.y;
      xs[r * Dm + c0 + c + 2] = (__bf16)f.z;
      xs[r * Dm + c0 + c + 3] = (__bf16)f.w;
    }
  }
  __syncthreads();

  int wave = tid >> 5;
  int lane = tid & 31;
  int half = lane >> 4;   // 0: lanes 0-15, 1: lanes 16-31
  int ln   = lane & 15;

  const float* W1e = W1 + (size_t)e * Dm * FFm;
  const float* W2e = W2 + (size_t)e * FFm * Dm;

  v8f yacc[16];           // rows 0-15 in yacc[0..7], rows 16-31 in yacc[8..15]
#pragma unroll
  for (int t = 0; t < 16; ++t) yacc[t] = (v8f){};

#pragma unroll 1
  for (int p = 0; p < FFm / 128; ++p) {
    // ---- GEMM1: wave computes 32x16 h tile for cols n1..n1+15 of chunk p ----
    int n1 = p * 128 + wave * 16 + ln;
    v8f c0 = (v8f){};     // rows 0-15
    v8f c1 = (v8f){};     // rows 16-31
#pragma unroll 1
    for (int ks = 0; ks < Dm / 32; ++ks) {
      int k0 = ks * 32;
      int koff = k0 + half * 16;
      // A fragments (16x32 bf16): lane holds row, K = koff+[0..15]
      union { uint4 u[2]; v16bf v; } A0, A1;
      A0.u[0] = *(const uint4*)(xs + ln * Dm + koff);
      A0.u[1] = *(const uint4*)(xs + ln * Dm + koff + 8);
      A1.u[0] = *(const uint4*)(xs + (16 + ln) * Dm + koff);
      A1.u[1] = *(const uint4*)(xs + (16 + ln) * Dm + koff + 8);
      // B fragment (32x16): lane holds col=n1, K = k0 + half*16 + [0..15]
      v16bf Bf;
      const float* wp = W1e + (size_t)(k0 + half * 16) * FFm + n1;
#pragma unroll
      for (int i = 0; i < 16; ++i) Bf[i] = (__bf16)wp[(size_t)i * FFm];
      c0 = __builtin_amdgcn_wmma_f32_16x16x32_bf16(false, A0.v, false, Bf,
                                                   (short)0, c0, false, false);
      c1 = __builtin_amdgcn_wmma_f32_16x16x32_bf16(false, A1.v, false, Bf,
                                                   (short)0, c1, false, false);
    }
    float bias1 = b1[e * FFm + n1];
    __syncthreads();   // all waves done reading hs from previous pass
    // bias + exact GELU, store bf16 h tiles. C layout: VGPR j -> M=j+half*8, N=ln
#pragma unroll
    for (int j = 0; j < 8; ++j) {
      float v = c0[j] + bias1;
      v = 0.5f * v * (1.0f + erff(v * 0.70710678118654752f));
      hs[(j + half * 8) * 128 + wave * 16 + ln] = (__bf16)v;
      float w = c1[j] + bias1;
      w = 0.5f * w * (1.0f + erff(w * 0.70710678118654752f));
      hs[(16 + j + half * 8) * 128 + wave * 16 + ln] = (__bf16)w;
    }
    __syncthreads();

    // ---- GEMM2: accumulate y; this wave owns output cols [wave*128, +128) ----
#pragma unroll 1
    for (int ks = 0; ks < 4; ++ks) {
      int k0 = ks * 32;
      int koff = k0 + half * 16;
      union { uint4 u[2]; v16bf v; } A0, A1;
      A0.u[0] = *(const uint4*)(hs + ln * 128 + koff);
      A0.u[1] = *(const uint4*)(hs + ln * 128 + koff + 8);
      A1.u[0] = *(const uint4*)(hs + (16 + ln) * 128 + koff);
      A1.u[1] = *(const uint4*)(hs + (16 + ln) * 128 + koff + 8);
      int kg = p * 128 + k0 + half * 16;
      const float* wbase = W2e + (size_t)kg * Dm + wave * 128 + ln;
#pragma unroll
      for (int t = 0; t < 8; ++t) {
        v16bf Bf;
        const float* wp = wbase + t * 16;
#pragma unroll
        for (int i = 0; i < 16; ++i) Bf[i] = (__bf16)wp[(size_t)i * Dm];
        yacc[t]     = __builtin_amdgcn_wmma_f32_16x16x32_bf16(false, A0.v, false, Bf,
                                                              (short)0, yacc[t], false, false);
        yacc[8 + t] = __builtin_amdgcn_wmma_f32_16x16x32_bf16(false, A1.v, false, Bf,
                                                              (short)0, yacc[8 + t], false, false);
      }
    }
  }

  // Epilogue: y + b2 -> per-(token,slot) partial output (unweighted, fp32)
#pragma unroll
  for (int t = 0; t < 8; ++t) {
    int n = wave * 128 + t * 16 + ln;
    float bias2 = b2[e * Dm + n];
#pragma unroll
    for (int j = 0; j < 8; ++j) {
      int m = j + half * 8;
      if (s_valid[m]) {
        yws[((size_t)s_tok[m] * Km + s_slot[m]) * Dm + n] = yacc[t][j] + bias2;
      }
      int m2 = 16 + m;
      if (s_valid[m2]) {
        yws[((size_t)s_tok[m2] * Km + s_slot[m2]) * Dm + n] = yacc[8 + t][j] + bias2;
      }
    }
  }
}

// ---------------------------------------------------------------------------
// 5) Combine: out[n] = w0 * y[n,slot0] + w1 * y[n,slot1]
// ---------------------------------------------------------------------------
__global__ void moe_combine(const float* __restrict__ yws,
                            const float* __restrict__ wts,
                            float* __restrict__ out) {
  int i = blockIdx.x * blockDim.x + threadIdx.x;  // over N*D/4
  if (i >= Nm * Dm / 4) return;
  int n  = i / (Dm / 4);
  int d4 = (i % (Dm / 4)) * 4;
  float w0 = wts[n * 2 + 0], w1 = wts[n * 2 + 1];
  float4 y0 = *(const float4*)(yws + ((size_t)n * Km + 0) * Dm + d4);
  float4 y1 = *(const float4*)(yws + ((size_t)n * Km + 1) * Dm + d4);
  float4 o;
  o.x = w0 * y0.x + w1 * y1.x;
  o.y = w0 * y0.y + w1 * y1.y;
  o.z = w0 * y0.z + w1 * y1.z;
  o.w = w0 * y0.w + w1 * y1.w;
  *(float4*)(out + (size_t)n * Dm + d4) = o;
}

// ---------------------------------------------------------------------------
extern "C" void kernel_launch(void* const* d_in, const int* in_sizes, int n_in,
                              void* d_out, int out_size, void* d_ws, size_t ws_size,
                              hipStream_t stream) {
  (void)in_sizes; (void)n_in; (void)out_size; (void)ws_size;
  const float* x  = (const float*)d_in[0];
  const float* Wr = (const float*)d_in[1];
  const float* W1 = (const float*)d_in[2];
  const float* b1 = (const float*)d_in[3];
  const float* W2 = (const float*)d_in[4];
  const float* b2 = (const float*)d_in[5];
  float* out = (float*)d_out;

  // workspace layout
  char* ws = (char*)d_ws;
  int*   idx    = (int*)  (ws);                                   // N*K ints
  float* wts    = (float*)(ws + (size_t)Nm * Km * 4);             // N*K floats
  int*   counts = (int*)  (ws + (size_t)2 * Nm * Km * 4);         // E ints (pad 256)
  int*   lists  = (int*)  (ws + (size_t)2 * Nm * Km * 4 + 256);   // E*N ints
  float* yws    = (float*)(ws + (size_t)2 * Nm * Km * 4 + 256 +
                                (size_t)Em * Nm * 4);             // N*K*D floats

  moe_router     <<<Nm / 8, 256, 0, stream>>>(x, Wr, idx, wts);
  moe_zero_counts<<<1, 64, 0, stream>>>(counts);
  moe_scatter    <<<(Nm * Km + 255) / 256, 256, 0, stream>>>(idx, counts, lists);
  moe_expert_ffn <<<Em * MAX_TILES, 256, 0, stream>>>(x, W1, b1, W2, b2,
                                                      counts, lists, yws);
  moe_combine    <<<(Nm * Dm / 4 + 255) / 256, 256, 0, stream>>>(yws, wts, out);
}
